// SwitchSparseMoELayer_68272800137427
// MI455X (gfx1250) — matrix operand order repaired and checked by reading, not compiled, not run on hardware
//
#include <hip/hip_runtime.h>

#define T_TOK 4096
#define DDIM  1024
#define IDIM  2048
#define NEXP  8

typedef __bf16 bf16_t;
typedef __attribute__((ext_vector_type(16))) __bf16 v16bf;
typedef __attribute__((ext_vector_type(8)))  __bf16 v8bf;
typedef __attribute__((ext_vector_type(4)))  __bf16 v4bf;
typedef __attribute__((ext_vector_type(8)))  float  v8f;

__device__ __forceinline__ v8f wmma_bf16(v16bf a, v16bf b, v8f c) {
  return __builtin_amdgcn_wmma_f32_16x16x32_bf16(false, a, false, b, (short)0, c,
                                                 false, false);
}

// gfx1250 async global->LDS copy (16B per lane), tracked by ASYNCcnt.
__device__ __forceinline__ void async_load_b128(unsigned lds_off, const void* gaddr) {
  asm volatile("global_load_async_to_lds_b128 %0, %1, off"
               :: "v"(lds_off), "v"(gaddr) : "memory");
}
__device__ __forceinline__ void wait_async() {
  asm volatile("s_wait_asynccnt 0x0" ::: "memory");
}

__device__ __forceinline__ v8bf cvt8(float4 a, float4 b) {
  v8bf r = { (bf16_t)a.x, (bf16_t)a.y, (bf16_t)a.z, (bf16_t)a.w,
             (bf16_t)b.x, (bf16_t)b.y, (bf16_t)b.z, (bf16_t)b.w };
  return r;
}

// ---------------------------------------------------------------- router
__global__ __launch_bounds__(128) void k_router(
    const float* __restrict__ x, const float* __restrict__ gw,
    const float* __restrict__ pb, const float* __restrict__ tf,
    int* __restrict__ sel, float* __restrict__ rw,
    float* __restrict__ lsq, float* __restrict__ probs)
{
  const int t = blockIdx.x;
  const float* xr = x + (size_t)t * DDIM;
  float acc[NEXP];
#pragma unroll
  for (int e = 0; e < NEXP; ++e) acc[e] = 0.f;
  for (int d = threadIdx.x; d < DDIM; d += 128) {
    float xv = xr[d];
    const float4* g = (const float4*)(gw + (size_t)d * NEXP);
    float4 g0 = g[0], g1 = g[1];
    acc[0] += xv * g0.x; acc[1] += xv * g0.y; acc[2] += xv * g0.z; acc[3] += xv * g0.w;
    acc[4] += xv * g1.x; acc[5] += xv * g1.y; acc[6] += xv * g1.z; acc[7] += xv * g1.w;
  }
#pragma unroll
  for (int e = 0; e < NEXP; ++e)
#pragma unroll
    for (int o = 16; o > 0; o >>= 1) acc[e] += __shfl_xor(acc[e], o, 32);

  __shared__ float red[4][NEXP];
  const int wid = threadIdx.x >> 5, lane = threadIdx.x & 31;
  if (lane == 0)
#pragma unroll
    for (int e = 0; e < NEXP; ++e) red[wid][e] = acc[e];
  __syncthreads();
  if (threadIdx.x == 0) {
    float lg[NEXP];
#pragma unroll
    for (int e = 0; e < NEXP; ++e) {
      float v = red[0][e] + red[1][e] + red[2][e] + red[3][e] + pb[e];
      float tp = fminf(fmaxf(tf[e], 0.1f), 10.0f);
      lg[e] = v / tp;
    }
    float mx = lg[0]; int am = 0;
#pragma unroll
    for (int e = 1; e < NEXP; ++e) if (lg[e] > mx) { mx = lg[e]; am = e; }
    float pr[NEXP]; float se = 0.f;
#pragma unroll
    for (int e = 0; e < NEXP; ++e) { pr[e] = __expf(lg[e] - mx); se += pr[e]; }
    const float inv = 1.f / se;
#pragma unroll
    for (int e = 0; e < NEXP; ++e) probs[(size_t)t * NEXP + e] = pr[e] * inv;
    sel[t] = am;
    rw[t]  = pr[am] * inv;
    float lse = mx + __logf(se);
    lsq[t] = lse * lse;
  }
}

// -------------------------------------------------------------- finalize
__global__ __launch_bounds__(1024) void k_finalize(
    const int* __restrict__ sel, const float* __restrict__ lsq,
    const float* __restrict__ probs,
    int* __restrict__ offs, int* __restrict__ cursor,
    float* __restrict__ aux_out)
{
  __shared__ float sred[1024];
  __shared__ int   scnt[1024];
  const int tid = threadIdx.x;
  float ps[NEXP]; int cn[NEXP];
#pragma unroll
  for (int e = 0; e < NEXP; ++e) { ps[e] = 0.f; cn[e] = 0; }
  float z = 0.f;
  for (int t = tid; t < T_TOK; t += 1024) {
    cn[sel[t]]++;
    z += lsq[t];
#pragma unroll
    for (int e = 0; e < NEXP; ++e) ps[e] += probs[(size_t)t * NEXP + e];
  }
  sred[tid] = z; __syncthreads();
  for (int s = 512; s > 0; s >>= 1) { if (tid < s) sred[tid] += sred[tid + s]; __syncthreads(); }
  const float ztot = sred[0]; __syncthreads();

  float lb = 0.f;
  int counts[NEXP];
  for (int e = 0; e < NEXP; ++e) {
    sred[tid] = ps[e]; scnt[tid] = cn[e]; __syncthreads();
    for (int s = 512; s > 0; s >>= 1) {
      if (tid < s) { sred[tid] += sred[tid + s]; scnt[tid] += scnt[tid + s]; }
      __syncthreads();
    }
    if (tid == 0) {
      counts[e] = scnt[0];
      lb += ((float)scnt[0] / T_TOK) * (sred[0] / T_TOK);
    }
    __syncthreads();
  }
  if (tid < NEXP) cursor[tid] = 0;
  if (tid == 0) {
    int run = 0;
    for (int e = 0; e < NEXP; ++e) { offs[e] = run; run += counts[e]; }
    offs[NEXP] = run;
    aux_out[0] = 0.01f * ((float)NEXP * lb) + 0.001f * (ztot / T_TOK);
  }
}

// ------------------------------------------------------- scatter / gather
__global__ void k_scatter(const int* __restrict__ sel, const int* __restrict__ offs,
                          int* __restrict__ cursor, int* __restrict__ perm)
{
  int t = blockIdx.x * blockDim.x + threadIdx.x;
  if (t < T_TOK) {
    int e = sel[t];
    int pos = offs[e] + atomicAdd(&cursor[e], 1);
    perm[pos] = t;
  }
}

__global__ __launch_bounds__(256) void k_gather(
    const float* __restrict__ x, const int* __restrict__ perm,
    const float* __restrict__ rw, bf16_t* __restrict__ Xg,
    float* __restrict__ rws)
{
  const int pos = blockIdx.x;
  const int t = perm[pos];
  const float4* src = (const float4*)(x + (size_t)t * DDIM);
  v4bf* dst = (v4bf*)(Xg + (size_t)pos * DDIM);
  float4 v = src[threadIdx.x];
  v4bf o = { (bf16_t)v.x, (bf16_t)v.y, (bf16_t)v.z, (bf16_t)v.w };
  dst[threadIdx.x] = o;
  if (threadIdx.x == 0) rws[pos] = rw[t];
}

// ------------------------------------------------------------- FFN GEMM 1
// H = silu(X*Wg) .* (X*Wu), double-buffered LDS, async A tiles.
__global__ __launch_bounds__(256) void k_ffn1(
    const bf16_t* __restrict__ Xg,
    const float* __restrict__ wgate, const float* __restrict__ wup,
    const int* __restrict__ offs, bf16_t* __restrict__ Hmat)
{
  const int e = blockIdx.z;
  const int seg0 = offs[e];
  const int Ne = offs[e + 1] - seg0;
  const int mbase = blockIdx.y * 64;
  if (mbase >= Ne) return;
  const int nbase = blockIdx.x * 64;

  __shared__ alignas(16) bf16_t sA[2][64 * 32];
  __shared__ alignas(16) bf16_t sG[2][32 * 64];
  __shared__ alignas(16) bf16_t sU[2][32 * 64];

  const int tid = threadIdx.x;
  const int lane = tid & 31;
  const int wave = tid >> 5;
  const int wm = (wave >> 1) * 16;
  const int wn = (wave & 1) * 32;

  v8f aG0 = {0.f,0.f,0.f,0.f,0.f,0.f,0.f,0.f};
  v8f aG1 = aG0, aU0 = aG0, aU1 = aG0;

  // A tile: 16B per thread, OOB rows clamped (masked again in epilogue).
  const int ar = tid >> 2;
  const int ak = (tid & 3) * 8;
  const int arow = (mbase + ar < Ne) ? (mbase + ar) : (Ne - 1);
  const bf16_t* Ap = Xg + (size_t)(seg0 + arow) * DDIM + ak;
  const unsigned aoff[2] = { (unsigned)(size_t)&sA[0][ar * 32 + ak],
                             (unsigned)(size_t)&sA[1][ar * 32 + ak] };

  // B tiles: 8 fp32 per thread per panel, converted to bf16.
  const int br = tid >> 3;
  const int bc = (tid & 7) * 8;
  const float* Gp = wgate + ((size_t)e * DDIM + br) * IDIM + nbase + bc;
  const float* Up = wup   + ((size_t)e * DDIM + br) * IDIM + nbase + bc;

  const int am  = wm + (lane & 15);
  const int akb = (lane >> 4) * 8;
  const int bk  = lane;

  float4 gr0, gr1, ur0, ur1;
  // prologue: stage 0
  async_load_b128(aoff[0], (const void*)Ap);
  gr0 = *(const float4*)Gp; gr1 = *(const float4*)(Gp + 4);
  ur0 = *(const float4*)Up; ur1 = *(const float4*)(Up + 4);

  for (int k0 = 0, it = 0; k0 < DDIM; k0 += 32, ++it) {
    const int buf = it & 1;
    *(v8bf*)(&sG[buf][br * 64 + bc]) = cvt8(gr0, gr1);
    *(v8bf*)(&sU[buf][br * 64 + bc]) = cvt8(ur0, ur1);
    wait_async();
    __syncthreads();

    const int kn = k0 + 32;
    if (kn < DDIM) {
      async_load_b128(aoff[buf ^ 1], (const void*)(Ap + kn));
      const float* gp = Gp + (size_t)kn * IDIM;
      gr0 = *(const float4*)gp; gr1 = *(const float4*)(gp + 4);
      const float* up = Up + (size_t)kn * IDIM;
      ur0 = *(const float4*)up; ur1 = *(const float4*)(up + 4);
    }

    union Frag { v16bf v; v8bf h[2]; };
    Frag a, g0f, g1f, u0f, u1f;
    a.h[0] = *(const v8bf*)(&sA[buf][am * 32 + akb]);
    a.h[1] = *(const v8bf*)(&sA[buf][am * 32 + akb + 16]);
#pragma unroll
    for (int s = 0; s < 16; ++s) {
      const int kk = (s < 8) ? (akb + s) : (akb + 8 + s);
      const int bn = wn + (lane & 15);
      g0f.v[s] = sG[buf][kk * 64 + bn];
      g1f.v[s] = sG[buf][kk * 64 + bn + 16];
      u0f.v[s] = sU[buf][kk * 64 + bn];
      u1f.v[s] = sU[buf][kk * 64 + bn + 16];
    }

    aG0 = wmma_bf16(a.v, g0f.v, aG0);
    aG1 = wmma_bf16(a.v, g1f.v, aG1);
    aU0 = wmma_bf16(a.v, u0f.v, aU0);
    aU1 = wmma_bf16(a.v, u1f.v, aU1);
    // single barrier per iteration: buffer parity guarantees no WAR on LDS
  }

  const int cn = lane & 15;
  const int cm = (lane >> 4) * 8;
#pragma unroll
  for (int r = 0; r < 8; ++r) {
    const int m = mbase + wm + cm + r;
    if (m < Ne) {
      bf16_t* hrow = Hmat + (size_t)(seg0 + m) * IDIM + nbase + wn;
      float g = aG0[r];
      float h = g * __builtin_amdgcn_rcpf(1.f + __expf(-g)) * aU0[r];
      hrow[cn] = (bf16_t)h;
      g = aG1[r];
      h = g * __builtin_amdgcn_rcpf(1.f + __expf(-g)) * aU1[r];
      hrow[16 + cn] = (bf16_t)h;
    }
  }
}

// ------------------------------------------------------------- FFN GEMM 2
// Out = (H @ Wd) * routing_weight, scattered to original token rows.
__global__ __launch_bounds__(256) void k_ffn2(
    const bf16_t* __restrict__ Hmat, const float* __restrict__ wdown,
    const int* __restrict__ offs, const int* __restrict__ perm,
    const float* __restrict__ rws, float* __restrict__ out)
{
  const int e = blockIdx.z;
  const int seg0 = offs[e];
  const int Ne = offs[e + 1] - seg0;
  const int mbase = blockIdx.y * 64;
  if (mbase >= Ne) return;
  const int nbase = blockIdx.x * 64;

  __shared__ alignas(16) bf16_t sA[2][64 * 32];
  __shared__ alignas(16) bf16_t sB[2][32 * 64];

  const int tid = threadIdx.x;
  const int lane = tid & 31;
  const int wave = tid >> 5;
  const int wm = (wave >> 1) * 16;
  const int wn = (wave & 1) * 32;

  v8f a0 = {0.f,0.f,0.f,0.f,0.f,0.f,0.f,0.f};
  v8f a1 = a0;

  const int ar = tid >> 2;
  const int ak = (tid & 3) * 8;
  const int arow = (mbase + ar < Ne) ? (mbase + ar) : (Ne - 1);
  const bf16_t* Ap = Hmat + (size_t)(seg0 + arow) * IDIM + ak;
  const unsigned aoff[2] = { (unsigned)(size_t)&sA[0][ar * 32 + ak],
                             (unsigned)(size_t)&sA[1][ar * 32 + ak] };

  const int br = tid >> 3;
  const int bc = (tid & 7) * 8;
  const float* Bp = wdown + ((size_t)e * IDIM + br) * DDIM + nbase + bc;

  const int am  = wm + (lane & 15);
  const int akb = (lane >> 4) * 8;
  const int bk  = lane;

  float4 br0, br1;
  async_load_b128(aoff[0], (const void*)Ap);
  br0 = *(const float4*)Bp; br1 = *(const float4*)(Bp + 4);

  for (int k0 = 0, it = 0; k0 < IDIM; k0 += 32, ++it) {
    const int buf = it & 1;
    *(v8bf*)(&sB[buf][br * 64 + bc]) = cvt8(br0, br1);
    wait_async();
    __syncthreads();

    const int kn = k0 + 32;
    if (kn < IDIM) {
      async_load_b128(aoff[buf ^ 1], (const void*)(Ap + kn));
      const float* bp = Bp + (size_t)kn * DDIM;
      br0 = *(const float4*)bp; br1 = *(const float4*)(bp + 4);
    }

    union Frag { v16bf v; v8bf h[2]; };
    Frag a, f0, f1;
    a.h[0] = *(const v8bf*)(&sA[buf][am * 32 + akb]);
    a.h[1] = *(const v8bf*)(&sA[buf][am * 32 + akb + 16]);
#pragma unroll
    for (int s = 0; s < 16; ++s) {
      const int kk = (s < 8) ? (akb + s) : (akb + 8 + s);
      const int bn = wn + (lane & 15);
      f0.v[s] = sB[buf][kk * 64 + bn];
      f1.v[s] = sB[buf][kk * 64 + bn + 16];
    }

    a0 = wmma_bf16(a.v, f0.v, a0);
    a1 = wmma_bf16(a.v, f1.v, a1);
  }

  const int cn = lane & 15;
  const int cm = (lane >> 4) * 8;
#pragma unroll
  for (int r = 0; r < 8; ++r) {
    const int m = mbase + wm + cm + r;
    if (m < Ne) {
      const int pos = seg0 + m;
      const int t = perm[pos];
      const float w = rws[pos];
      float* orow = out + (size_t)t * DDIM + nbase + wn;
      orow[cn]      = a0[r] * w;
      orow[16 + cn] = a1[r] * w;
    }
  }
}

// ---------------------------------------------------------------- launch
extern "C" void kernel_launch(void* const* d_in, const int* in_sizes, int n_in,
                              void* d_out, int out_size, void* d_ws, size_t ws_size,
                              hipStream_t stream)
{
  (void)in_sizes; (void)n_in; (void)out_size; (void)ws_size;
  const float* x  = (const float*)d_in[0];
  const float* gw = (const float*)d_in[1];
  const float* wg = (const float*)d_in[2];
  const float* wu = (const float*)d_in[3];
  const float* wd = (const float*)d_in[4];
  const float* pb = (const float*)d_in[5];
  const float* tf = (const float*)d_in[6];
  float* out = (float*)d_out;

  char* ws = (char*)d_ws;
  int*    offs   = (int*)(ws + 0);                                   // 9 ints
  int*    cursor = (int*)(ws + 256);                                 // 8 ints
  int*    sel    = (int*)(ws + 1024);                                // T ints
  float*  rw     = (float*)(ws + 1024 + 4 * T_TOK);                  // T floats
  float*  lsq    = (float*)(ws + 1024 + 8 * T_TOK);                  // T floats
  float*  probs  = (float*)(ws + 1024 + 12 * T_TOK);                 // T*E floats
  int*    perm   = (int*)(ws + 1024 + 12 * T_TOK + 4 * T_TOK * NEXP);
  float*  rws    = (float*)(ws + 1024 + 16 * T_TOK + 4 * T_TOK * NEXP);
  bf16_t* Xg     = (bf16_t*)(ws + 1024 + 20 * T_TOK + 4 * T_TOK * NEXP);
  bf16_t* Hmat   = (bf16_t*)(ws + 1024 + 20 * T_TOK + 4 * T_TOK * NEXP
                             + 2 * (size_t)T_TOK * DDIM);

  k_router<<<T_TOK, 128, 0, stream>>>(x, gw, pb, tf, sel, rw, lsq, probs);
  k_finalize<<<1, 1024, 0, stream>>>(sel, lsq, probs, offs, cursor,
                                     out + (size_t)T_TOK * DDIM);
  k_scatter<<<T_TOK / 256, 256, 0, stream>>>(sel, offs, cursor, perm);
  k_gather<<<T_TOK, 256, 0, stream>>>(x, perm, rw, Xg, rws);

  dim3 g1(IDIM / 64, T_TOK / 64, NEXP);
  k_ffn1<<<g1, 256, 0, stream>>>(Xg, wg, wu, offs, Hmat);
  dim3 g2(DDIM / 64, T_TOK / 64, NEXP);
  k_ffn2<<<g2, 256, 0, stream>>>(Hmat, wd, offs, perm, rws, out);
}
